// RBLNGptOssMLP_46231027974604
// MI455X (gfx1250) — matrix operand order repaired and checked by reading, not compiled, not run on hardware
//
#include <hip/hip_runtime.h>
#include <hip/hip_bf16.h>
#include <cstdint>

#define ALPHA_F 1.702f
#define LIMIT_F 7.0f
#define T_TOK 1024
#define H_DIM 2048
#define N_EXP 16
#define D_DIM 2048
#define TWO_D 4096

typedef __attribute__((ext_vector_type(16))) __bf16 v16bf;
typedef __attribute__((ext_vector_type(8)))  float  v8f;

// --- CDNA5 async global->LDS copy (ASYNCcnt-tracked), per-lane addresses ----
__device__ __forceinline__ void async_ld_b128(uint32_t lds_off, const void* g) {
  asm volatile("global_load_async_to_lds_b128 %0, %1, off"
               :: "v"(lds_off), "v"((unsigned long long)(uintptr_t)g)
               : "memory");
}
__device__ __forceinline__ void wait_async0() {
  asm volatile("s_wait_asynccnt 0x0" ::: "memory");
}

// ---------------------------------------------------------------------------
// Kernel 1: router. One wave32 per token: logits = x @ Wr^T + b, top-4 softmax,
// scatter (token, score) into per-expert compacted lists via atomics.
// ---------------------------------------------------------------------------
__global__ void __launch_bounds__(128) moe_router_kernel(
    const float* __restrict__ x, const float* __restrict__ rw,
    const float* __restrict__ rb, int* __restrict__ cnt,
    int* __restrict__ tlist, float* __restrict__ tscore)
{
  const int wave = threadIdx.x >> 5;
  const int lane = threadIdx.x & 31;
  const int t = blockIdx.x * 4 + wave;           // grid is exactly T/4 blocks

  float acc[N_EXP];
#pragma unroll
  for (int e = 0; e < N_EXP; ++e) acc[e] = 0.f;

  const float* xp = x + (size_t)t * H_DIM;
  for (int k = lane; k < H_DIM; k += 32) {
    const float xv = xp[k];
#pragma unroll
    for (int e = 0; e < N_EXP; ++e) acc[e] += xv * rw[(size_t)e * H_DIM + k];
  }
#pragma unroll
  for (int e = 0; e < N_EXP; ++e) {
#pragma unroll
    for (int off = 16; off > 0; off >>= 1) acc[e] += __shfl_xor(acc[e], off, 32);
  }

  if (lane == 0) {
    float lg[N_EXP];
#pragma unroll
    for (int e = 0; e < N_EXP; ++e) lg[e] = acc[e] + rb[e];
    int   idx[4];
    float val[4];
#pragma unroll
    for (int s = 0; s < 4; ++s) {        // top-4, first index wins ties
      int bi = 0; float bv = -3.4e38f;
      for (int e = 0; e < N_EXP; ++e) { if (lg[e] > bv) { bv = lg[e]; bi = e; } }
      idx[s] = bi; val[s] = bv; lg[bi] = -3.4e38f;
    }
    const float mx = val[0];
    float ex[4], se = 0.f;
#pragma unroll
    for (int s = 0; s < 4; ++s) { ex[s] = __expf(val[s] - mx); se += ex[s]; }
    const float inv = 1.f / se;
    for (int s = 0; s < 4; ++s) {
      const int e   = idx[s];
      const int pos = atomicAdd(&cnt[e], 1);
      tlist[e * T_TOK + pos]  = t;
      tscore[e * T_TOK + pos] = ex[s] * inv;
    }
  }
}

// ---------------------------------------------------------------------------
// Kernel 2: gate_up GEMM + fused clamp/GLU. Block = 32 tokens x 32 pair-cols.
// 4 waves: (pc_blk, role); role 0 = gate (even interleaved cols), role 1 = up.
// Each wave keeps TWO 16x16 accumulators (row sub-tiles) sharing one B frag.
// A tile is double-buffered in LDS via global_load_async_to_lds_b128.
// ---------------------------------------------------------------------------
__global__ void __launch_bounds__(128) moe_gateup_kernel(
    const float* __restrict__ x,
    const float* __restrict__ gup,     // [E][H][2D]
    const float* __restrict__ gupb,    // [E][2D]
    const int*   __restrict__ cnt,
    const int*   __restrict__ tlist,
    __bf16*      __restrict__ act)     // [E][T][D]
{
  const int e  = blockIdx.z;
  const int mt = blockIdx.y;           // 32-token tile
  const int n_tok = cnt[e];
  if (mt * 32 >= n_tok) return;        // uniform exit: EXEC all-ones for WMMA

  const int wid    = threadIdx.x >> 5;
  const int lane   = threadIdx.x & 31;
  const int pc_blk = wid >> 1;
  const int role   = wid & 1;
  const int m  = lane & 15;
  const int hi = lane >> 4;
  const int pc  = blockIdx.x * 32 + pc_blk * 16 + (lane & 15);
  const int col = 2 * pc + role;

  __shared__ __align__(16) float Asf[2][32][36];   // stride 144B: 16B-aligned, bank-spread
  __shared__ float upbuf[2][2][8][32];

  // staging map: 128 threads -> 32 rows x 4 groups of 8 floats (2x b128 each)
  const int srow = threadIdx.x >> 2;
  const int scg  = (threadIdx.x & 3) << 3;
  const int grow = mt * 32 + srow;
  const int tok  = (grow < n_tok) ? tlist[e * T_TOK + grow] : 0;  // safe fallback row
  const float* xrow = x + (size_t)tok * H_DIM + scg;

  const uint32_t l0 = (uint32_t)(uintptr_t)&Asf[0][srow][scg];
  const uint32_t l1 = (uint32_t)(uintptr_t)&Asf[1][srow][scg];

  async_ld_b128(l0,      xrow);        // prologue: stage k0 = 0 into buffer 0
  async_ld_b128(l0 + 16, xrow + 4);

  const float* Bbase = gup + ((size_t)e * H_DIM) * TWO_D + col;

  v8f acc0 = {}, acc1 = {};
  int p = 0;
  for (int k0 = 0; k0 < H_DIM; k0 += 32) {
    wait_async0();
    __syncthreads();
    if (k0 + 32 < H_DIM) {             // issue next A slab into alternate buffer
      const float* nx = xrow + k0 + 32;
      const uint32_t ln = p ? l0 : l1;
      async_ld_b128(ln,      nx);
      async_ld_b128(ln + 16, nx + 4);
    }

    v16bf a0, a1, b;
#pragma unroll
    for (int i = 0; i < 16; ++i) {     // ISA 16-bit A 16x32 fragment layout
      const int kk = ((i >> 3) << 4) + hi * 8 + (((i >> 1) & 3) << 1) + (i & 1);
      a0[i] = (__bf16)Asf[p][m][kk];
      a1[i] = (__bf16)Asf[p][16 + m][kk];
    }
    const float* bp = Bbase + (size_t)k0 * TWO_D;
    if (k0 + 32 < H_DIM)
      __builtin_prefetch((const void*)(bp + (size_t)32 * TWO_D), 0, 1);
#pragma unroll
    for (int i = 0; i < 16; ++i)       // ISA 16-bit B 32x16 fragment layout
      b[i] = (__bf16)bp[(size_t)(((i >> 3) << 4) + hi * 8 + (i & 7)) * TWO_D];

    acc0 = __builtin_amdgcn_wmma_f32_16x16x32_bf16(false, a0, false, b,
                                                   (short)0, acc0, false, false);
    acc1 = __builtin_amdgcn_wmma_f32_16x16x32_bf16(false, a1, false, b,
                                                   (short)0, acc1, false, false);
    p ^= 1;
  }

  const float bias = gupb[(size_t)e * TWO_D + col];
  if (role == 1) {                     // up: clamp to [-LIMIT, LIMIT]
#pragma unroll
    for (int r = 0; r < 8; ++r) {
      upbuf[0][pc_blk][r][lane] = fminf(fmaxf(acc0[r] + bias, -LIMIT_F), LIMIT_F);
      upbuf[1][pc_blk][r][lane] = fminf(fmaxf(acc1[r] + bias, -LIMIT_F), LIMIT_F);
    }
  }
  __syncthreads();
  if (role == 0) {                     // gate: min(LIMIT), GLU, combine, store bf16
#pragma unroll
    for (int r = 0; r < 8; ++r) {
      const float g0 = fminf(acc0[r] + bias, LIMIT_F);
      const float g1 = fminf(acc1[r] + bias, LIMIT_F);
      const float glu0 = g0 * (1.f / (1.f + __expf(-g0 * ALPHA_F)));
      const float glu1 = g1 * (1.f / (1.f + __expf(-g1 * ALPHA_F)));
      const float h0 = (upbuf[0][pc_blk][r][lane] + 1.f) * glu0;
      const float h1 = (upbuf[1][pc_blk][r][lane] + 1.f) * glu1;
      const int row0 = mt * 32 + r + hi * 8;       // C/D layout: m = r + hi*8
      act[((size_t)e * T_TOK + row0) * D_DIM + pc]        = (__bf16)h0;
      act[((size_t)e * T_TOK + row0 + 16) * D_DIM + pc]   = (__bf16)h1;
    }
  }
}

// ---------------------------------------------------------------------------
// Kernel 3: down GEMM + scaled scatter-add. Block = 32 tokens x 64 H-cols,
// 4 waves of 16 cols, two row sub-tiles per wave sharing one B fragment.
// bf16 A tile double-buffered via async-to-LDS (pure copy, no conversion).
// ---------------------------------------------------------------------------
__global__ void __launch_bounds__(128) moe_down_kernel(
    const __bf16* __restrict__ act,    // [E][T][D]
    const float*  __restrict__ dw,     // [E][D][H]
    const float*  __restrict__ dbias,  // [E][H]
    const int*    __restrict__ cnt,
    const int*    __restrict__ tlist,
    const float*  __restrict__ tscore,
    float*        __restrict__ out)    // [T][H], pre-zeroed
{
  const int e  = blockIdx.z;
  const int mt = blockIdx.y;           // 32-token tile
  const int n_tok = cnt[e];
  if (mt * 32 >= n_tok) return;

  const int wid  = threadIdx.x >> 5;
  const int lane = threadIdx.x & 31;
  const int m  = lane & 15;
  const int hi = lane >> 4;
  const int h  = blockIdx.x * 64 + wid * 16 + (lane & 15);

  __shared__ __align__(16) __bf16 Asb[2][32][40];  // stride 80B: 16B-aligned

  // staging map: 128 threads -> 32 rows x 4 groups of 8 bf16 (one b128 each)
  const int srow = threadIdx.x >> 2;
  const int scg  = (threadIdx.x & 3) << 3;
  const __bf16* arow = act + ((size_t)e * T_TOK + mt * 32 + srow) * D_DIM + scg;
  const uint32_t l0 = (uint32_t)(uintptr_t)&Asb[0][srow][scg];
  const uint32_t l1 = (uint32_t)(uintptr_t)&Asb[1][srow][scg];

  async_ld_b128(l0, arow);             // prologue: k0 = 0

  const float* Bbase = dw + ((size_t)e * D_DIM) * H_DIM + h;

  v8f acc0 = {}, acc1 = {};
  int p = 0;
  for (int k0 = 0; k0 < D_DIM; k0 += 32) {
    wait_async0();
    __syncthreads();
    if (k0 + 32 < D_DIM)
      async_ld_b128(p ? l0 : l1, arow + k0 + 32);

    v16bf a0, a1, b;
#pragma unroll
    for (int i = 0; i < 16; ++i) {
      const int kk = ((i >> 3) << 4) + hi * 8 + (((i >> 1) & 3) << 1) + (i & 1);
      a0[i] = Asb[p][m][kk];
      a1[i] = Asb[p][16 + m][kk];
    }
    const float* bp = Bbase + (size_t)k0 * H_DIM;
    if (k0 + 32 < D_DIM)
      __builtin_prefetch((const void*)(bp + (size_t)32 * H_DIM), 0, 1);
#pragma unroll
    for (int i = 0; i < 16; ++i)
      b[i] = (__bf16)bp[(size_t)(((i >> 3) << 4) + hi * 8 + (i & 7)) * H_DIM];

    acc0 = __builtin_amdgcn_wmma_f32_16x16x32_bf16(false, a0, false, b,
                                                   (short)0, acc0, false, false);
    acc1 = __builtin_amdgcn_wmma_f32_16x16x32_bf16(false, a1, false, b,
                                                   (short)0, acc1, false, false);
    p ^= 1;
  }

  const float bias = dbias[(size_t)e * H_DIM + h];
#pragma unroll
  for (int r = 0; r < 8; ++r) {
    const int g0 = mt * 32 + r + hi * 8;
    if (g0 < n_tok) {
      const int   tokv = tlist[e * T_TOK + g0];
      const float s    = tscore[e * T_TOK + g0];
      atomicAdd(&out[(size_t)tokv * H_DIM + h], s * (acc0[r] + bias));
    }
    const int g1 = g0 + 16;
    if (g1 < n_tok) {
      const int   tokv = tlist[e * T_TOK + g1];
      const float s    = tscore[e * T_TOK + g1];
      atomicAdd(&out[(size_t)tokv * H_DIM + h], s * (acc1[r] + bias));
    }
  }
}

// ---------------------------------------------------------------------------
extern "C" void kernel_launch(void* const* d_in, const int* in_sizes, int n_in,
                              void* d_out, int out_size, void* d_ws, size_t ws_size,
                              hipStream_t stream) {
  const float* x     = (const float*)d_in[0];  // [1,1024,2048]
  const float* rw    = (const float*)d_in[1];  // [16,2048]
  const float* rb    = (const float*)d_in[2];  // [16]
  const float* gup   = (const float*)d_in[3];  // [16,2048,4096]
  const float* gupb  = (const float*)d_in[4];  // [16,4096]
  const float* dw    = (const float*)d_in[5];  // [16,2048,2048]
  const float* dbias = (const float*)d_in[6];  // [16,2048]
  float* out = (float*)d_out;

  // Workspace layout
  uint8_t* ws = (uint8_t*)d_ws;
  int*    cnt    = (int*)ws;                                        // 256 B slot
  int*    tlist  = (int*)(ws + 256);                                // E*T ints
  float*  tscore = (float*)(ws + 256 + (size_t)N_EXP * T_TOK * 4);  // E*T floats
  __bf16* act    = (__bf16*)(ws + 256 + 2 * (size_t)N_EXP * T_TOK * 4); // E*T*D bf16

  hipMemsetAsync(cnt, 0, 256, stream);
  hipMemsetAsync(out, 0, (size_t)T_TOK * H_DIM * sizeof(float), stream);

  moe_router_kernel<<<T_TOK / 4, 128, 0, stream>>>(x, rw, rb, cnt, tlist, tscore);

  moe_gateup_kernel<<<dim3(D_DIM / 32, T_TOK / 32, N_EXP), 128, 0, stream>>>(
      x, gup, gupb, cnt, tlist, act);

  moe_down_kernel<<<dim3(H_DIM / 64, T_TOK / 32, N_EXP), 128, 0, stream>>>(
      act, dw, dbias, cnt, tlist, tscore, out);
}
